// Transformer_16174846837146
// MI455X (gfx1250) — compile-verified, matrix operand order checked by reading, not run on hardware
//
#include <hip/hip_runtime.h>
#include <hip/hip_bf16.h>

// ---------------------------------------------------------------------------
// DETR-style decoder layer for MI455X (gfx1250, wave32, WMMA).
// All dense contractions run on v_wmma_f32_16x16x32_f16 (f16 in, f32 acc).
// GEMM tiles are staged with CDNA5 async global->LDS copies (ASYNCcnt).
// Attention is flash-style (online softmax), Gaussian bias fused into CA
// logits so the 157MB [B,H,L,S] bias tensor is never materialized.
// ---------------------------------------------------------------------------

typedef __attribute__((ext_vector_type(16))) _Float16 v16h;
typedef __attribute__((ext_vector_type(8)))  float    v8f;
typedef __attribute__((ext_vector_type(8)))  _Float16 h8v;

#define Lq_  300
#define Bb_  16
#define Ss_  1024
#define Dd_  256
#define Hh_  8
#define HD_  32
#define FF_  2048
#define LB_  (Lq_*Bb_)   // 4800 tokens
#define SB_  (Ss_*Bb_)   // 16384 memory tokens

// ---- WMMA wrapper ----------------------------------------------------------
__device__ inline v8f wmma_f16(v16h a, v16h b, v8f c) {
  // D = A(16x32 f16) x B^T(16x32 f16, N-major) + C(16x16 f32)
  return __builtin_amdgcn_wmma_f32_16x16x32_f16(
      /*neg_a=*/false, a, /*neg_b=*/false, b,
      /*c_mod=*/(short)0, c, /*reuse_a=*/false, /*reuse_b=*/false);
}

// Async 16-byte global->LDS copy (CDNA5 Tensor/async path, ASYNCcnt).
// INST_OFFSET is added to both the global and the LDS address (ISA 08 §4.4).
__device__ inline void async_g2l_b128_o0(unsigned lds_off, const void* gaddr) {
  asm volatile("global_load_async_to_lds_b128 %0, %1, off"
               :: "v"(lds_off), "v"(gaddr) : "memory");
}
__device__ inline void async_g2l_b128_o16(unsigned lds_off, const void* gaddr) {
  asm volatile("global_load_async_to_lds_b128 %0, %1, off offset:16"
               :: "v"(lds_off), "v"(gaddr) : "memory");
}
__device__ inline void wait_async0() {
  asm volatile("s_wait_asynccnt 0x0" ::: "memory");
}

// Load a 16x32 f16 operand tile in WMMA A (or B^T) register layout.
// p points at element (row0, col0) of a row-major matrix, leading dim ld.
// Per ISA: lanes 0-15 -> rows 0-15 (halves j: K = j<8 ? j : j+8);
//          lanes 16-31 -> same rows, K offset +8.
// Rows >= rows_left are zero-filled.  Requires 16B-aligned rows (ld % 8 == 0).
__device__ inline v16h load_op_tile(const _Float16* __restrict__ p, int ld,
                                    int lane, int rows_left) {
  const int r   = lane & 15;
  const int khi = (lane >> 4) << 3;  // 0 or 8
  v16h out;
  if (r < rows_left) {
    const _Float16* q = p + (size_t)r * ld + khi;
    h8v lo = *(const h8v*)(q);
    h8v hi = *(const h8v*)(q + 16);
#pragma unroll
    for (int j = 0; j < 8; ++j) { out[j] = lo[j]; out[j + 8] = hi[j]; }
  } else {
#pragma unroll
    for (int j = 0; j < 16; ++j) out[j] = (_Float16)0.0f;
  }
  return out;
}

// B^T tile gathered from a column of V: lane -> n (output dim), halves -> K=s.
__device__ inline v16h load_vT_tile(const _Float16* __restrict__ vbase, int ld,
                                    int lane, int s0, int Slimit) {
  const int n   = lane & 15;
  const int khi = (lane >> 4) << 3;
  v16h out;
#pragma unroll
  for (int j = 0; j < 16; ++j) {
    int k = khi + (j < 8 ? j : j + 8);
    int s = s0 + k;
    out[j] = (s < Slimit) ? vbase[(size_t)s * ld + n] : (_Float16)0.0f;
  }
  return out;
}

__device__ inline float rowmax16(float x) {
  x = fmaxf(x, __shfl_xor(x, 1, 32));
  x = fmaxf(x, __shfl_xor(x, 2, 32));
  x = fmaxf(x, __shfl_xor(x, 4, 32));
  x = fmaxf(x, __shfl_xor(x, 8, 32));
  return x;
}
__device__ inline float rowsum16(float x) {
  x += __shfl_xor(x, 1, 32);
  x += __shfl_xor(x, 2, 32);
  x += __shfl_xor(x, 4, 32);
  x += __shfl_xor(x, 8, 32);
  return x;
}

// ---- General WMMA GEMM: C = act(A[MxK] * Bt[NxK]^T + bias) ------------------
// Block: 256 threads = 8 waves, tile 64(M) x 128(N), each wave owns 32x32.
// Tiles staged via global_load_async_to_lds_b128 (no VGPR round trip).
__global__ __launch_bounds__(256) void gemm_kernel(
    const _Float16* __restrict__ A, const _Float16* __restrict__ Bt,
    const float* __restrict__ bias, float* __restrict__ C32,
    _Float16* __restrict__ C16, int M, int N, int K, int relu) {
  __shared__ __align__(16) _Float16 As[64][32];
  __shared__ __align__(16) _Float16 Bs[128][32];
  const int lane = threadIdx.x & 31;
  const int wv   = threadIdx.x >> 5;
  const int m0   = blockIdx.y * 64;
  const int n0   = blockIdx.x * 128;
  const int wm   = (wv & 1) * 32;
  const int wn   = (wv >> 1) * 32;
  v8f acc00 = {}; v8f acc01 = {}; v8f acc10 = {}; v8f acc11 = {};

  const int arow = threadIdx.x >> 2;            // 0..63
  const int acol = (threadIdx.x & 3) * 8;       // 0,8,16,24
  const int brow = threadIdx.x >> 1;            // 0..127
  const int bcol = (threadIdx.x & 1) * 16;      // 0,16

  const bool a_ok = (m0 + arow) < M;
  const bool b_ok = (n0 + brow) < N;
  const unsigned lds_a = (unsigned)(uintptr_t)(&As[arow][acol]);
  const unsigned lds_b = (unsigned)(uintptr_t)(&Bs[brow][bcol]);
  const _Float16* agp = A  + (size_t)(m0 + arow) * K + acol;
  const _Float16* bgp = Bt + (size_t)(n0 + brow) * K + bcol;

  // zero-fill out-of-range rows once; those LDS cells are never overwritten
  if (!a_ok) { h8v z = {}; *(h8v*)(&As[arow][acol]) = z; }
  if (!b_ok) {
    h8v z = {};
    *(h8v*)(&Bs[brow][bcol])     = z;
    *(h8v*)(&Bs[brow][bcol + 8]) = z;
  }

  for (int k0 = 0; k0 < K; k0 += 32) {
    if (a_ok) async_g2l_b128_o0(lds_a, agp + k0);
    if (b_ok) {
      async_g2l_b128_o0 (lds_b, bgp + k0);
      async_g2l_b128_o16(lds_b, bgp + k0);
    }
    wait_async0();
    __syncthreads();
    v16h a0 = load_op_tile(&As[wm +  0][0], 32, lane, 16);
    v16h a1 = load_op_tile(&As[wm + 16][0], 32, lane, 16);
    v16h b0 = load_op_tile(&Bs[wn +  0][0], 32, lane, 16);
    v16h b1 = load_op_tile(&Bs[wn + 16][0], 32, lane, 16);
    acc00 = wmma_f16(a0, b0, acc00);
    acc01 = wmma_f16(a0, b1, acc01);
    acc10 = wmma_f16(a1, b0, acc10);
    acc11 = wmma_f16(a1, b1, acc11);
    __syncthreads();
  }

  const int hi = lane >> 4;
  const int nn = lane & 15;
#pragma unroll
  for (int tm = 0; tm < 2; ++tm) {
#pragma unroll
    for (int tn = 0; tn < 2; ++tn) {
      v8f accv = (tm == 0) ? (tn == 0 ? acc00 : acc01)
                           : (tn == 0 ? acc10 : acc11);
#pragma unroll
      for (int i = 0; i < 8; ++i) {
        int m = m0 + wm + tm * 16 + i + 8 * hi;
        int n = n0 + wn + tn * 16 + nn;
        if (m < M && n < N) {
          float v = accv[i];
          if (bias) v += bias[n];
          if (relu) v = fmaxf(v, 0.0f);
          if (C32) C32[(size_t)m * N + n] = v;
          if (C16) C16[(size_t)m * N + n] = (_Float16)v;
        }
      }
    }
  }
}

// ---- Flash attention (one 16-row query tile per wave, online softmax) -------
// Q/K/V stored [T,B,D] f16; head (b,h) slice has row stride B*D.
// WITH_BIAS: fuse DETR gaussian bias  -|dx^2*sx + dy^2*sy|/8  into logits.
template <int WITH_BIAS>
__global__ __launch_bounds__(256) void attn_kernel(
    const _Float16* __restrict__ Qh, const _Float16* __restrict__ Kh,
    const _Float16* __restrict__ Vh, _Float16* __restrict__ Oh,
    const float* __restrict__ point, const float* __restrict__ scalev,
    const float* __restrict__ grid, int Lq, int S) {
  __shared__ __align__(16) _Float16 Plds[8][16][32];  // per-wave P tile
  const int lane = threadIdx.x & 31;
  const int wv   = threadIdx.x >> 5;
  const int bh   = blockIdx.x;              // b*H + h
  const int b    = bh >> 3;
  const int h    = bh & 7;
  const int l0   = blockIdx.y * 128 + wv * 16;
  const int ldq  = Bb_ * Dd_;
  const float sm_scale = 0.17677669529663689f;  // HD^-0.5

  const _Float16* Qp = Qh + (size_t)l0 * ldq + b * Dd_ + h * HD_;
  v16h qa = load_op_tile(Qp, ldq, lane, Lq - l0);

  float px[8], py[8], sx[8], sy[8];
  if (WITH_BIAS) {
    const int hi = lane >> 4;
#pragma unroll
    for (int i = 0; i < 8; ++i) {
      int l = l0 + i + 8 * hi;
      if (l < Lq) {
        size_t o = ((size_t)l * (Bb_ * Hh_) + bh) * 2;
        px[i] = point[o];  py[i] = point[o + 1];
        sx[i] = scalev[o]; sy[i] = scalev[o + 1];
      } else { px[i] = py[i] = sx[i] = sy[i] = 0.0f; }
    }
  }

  v8f o0 = {}; v8f o1 = {};
  float mrow[8], lrow[8];
#pragma unroll
  for (int i = 0; i < 8; ++i) { mrow[i] = -3.0e38f; lrow[i] = 0.0f; }

  for (int s0 = 0; s0 < S; s0 += 32) {
    const _Float16* Kp = Kh + (size_t)s0 * ldq + b * Dd_ + h * HD_;
    v16h kb0 = load_op_tile(Kp, ldq, lane, S - s0);
    v16h kb1 = load_op_tile(Kp + (size_t)16 * ldq, ldq, lane, S - s0 - 16);
    v8f z = {};
    v8f c0 = wmma_f16(qa, kb0, z);
    v8f c1 = wmma_f16(qa, kb1, z);

    const int sA = s0 + (lane & 15);
    const int sB = sA + 16;
    float gx0 = 0, gy0 = 0, gx1 = 0, gy1 = 0;
    if (WITH_BIAS) {
      if (sA < S) { gx0 = grid[2 * sA]; gy0 = grid[2 * sA + 1]; }
      if (sB < S) { gx1 = grid[2 * sB]; gy1 = grid[2 * sB + 1]; }
    }
#pragma unroll
    for (int i = 0; i < 8; ++i) {
      float v0 = c0[i] * sm_scale;
      float v1 = c1[i] * sm_scale;
      if (WITH_BIAS) {
        float dx0 = px[i] - gx0, dy0 = py[i] - gy0;
        float dx1 = px[i] - gx1, dy1 = py[i] - gy1;
        v0 -= fabsf(dx0 * dx0 * sx[i] + dy0 * dy0 * sy[i]) * 0.125f;
        v1 -= fabsf(dx1 * dx1 * sx[i] + dy1 * dy1 * sy[i]) * 0.125f;
      }
      if (sA >= S) v0 = -3.0e38f;
      if (sB >= S) v1 = -3.0e38f;
      c0[i] = v0; c1[i] = v1;
    }
    // online softmax update (per output row; rows live in 16-lane halves)
#pragma unroll
    for (int i = 0; i < 8; ++i) {
      float cm   = rowmax16(fmaxf(c0[i], c1[i]));
      float mn   = fmaxf(mrow[i], cm);
      float corr = __expf(mrow[i] - mn);
      float p0   = __expf(c0[i] - mn);
      float p1   = __expf(c1[i] - mn);
      float rs   = rowsum16(p0 + p1);
      lrow[i] = lrow[i] * corr + rs;
      mrow[i] = mn;
      o0[i] *= corr; o1[i] *= corr;
      c0[i] = p0; c1[i] = p1;
    }
    // C-layout -> A-layout transpose through per-wave LDS tile
    {
      const int hi = lane >> 4;
      const int nn = lane & 15;
#pragma unroll
      for (int i = 0; i < 8; ++i) {
        Plds[wv][i + 8 * hi][nn]      = (_Float16)c0[i];
        Plds[wv][i + 8 * hi][nn + 16] = (_Float16)c1[i];
      }
    }
    // same-wave LDS RAW: split-counter wait is sufficient (wave32 lockstep)
    asm volatile("s_wait_dscnt 0x0" ::: "memory");
    v16h pa = load_op_tile(&Plds[wv][0][0], 32, lane, 16);
    const _Float16* Vp = Vh + (size_t)b * Dd_ + h * HD_;
    v16h vb0 = load_vT_tile(Vp,      ldq, lane, s0, S);
    v16h vb1 = load_vT_tile(Vp + 16, ldq, lane, s0, S);
    o0 = wmma_f16(pa, vb0, o0);
    o1 = wmma_f16(pa, vb1, o1);
  }

  const int hi = lane >> 4;
  const int nn = lane & 15;
#pragma unroll
  for (int i = 0; i < 8; ++i) {
    int l = l0 + i + 8 * hi;
    if (l < Lq) {
      float inv = lrow[i] > 0.0f ? 1.0f / lrow[i] : 0.0f;
      size_t o = (size_t)l * ldq + b * Dd_ + h * HD_ + nn;
      Oh[o]      = (_Float16)(o0[i] * inv);
      Oh[o + 16] = (_Float16)(o1[i] * inv);
    }
  }
}

// ---- Residual add + LayerNorm (one 256-wide token per block) ----------------
__global__ __launch_bounds__(256) void add_ln_kernel(
    const float* __restrict__ x, const float* __restrict__ r,
    const float* __restrict__ g, const float* __restrict__ beta,
    float* __restrict__ y32, _Float16* __restrict__ y16) {
  const int t = blockIdx.x;
  const int i = threadIdx.x;
  const size_t idx = (size_t)t * 256 + i;
  float v = x[idx] + r[idx];
  __shared__ float red[8];
  float s = v;
  s += __shfl_xor(s, 1, 32);  s += __shfl_xor(s, 2, 32);
  s += __shfl_xor(s, 4, 32);  s += __shfl_xor(s, 8, 32);
  s += __shfl_xor(s, 16, 32);
  if ((i & 31) == 0) red[i >> 5] = s;
  __syncthreads();
  float mean = 0.0f;
#pragma unroll
  for (int w = 0; w < 8; ++w) mean += red[w];
  mean *= (1.0f / 256.0f);
  __syncthreads();
  float d = v - mean;
  float q = d * d;
  q += __shfl_xor(q, 1, 32);  q += __shfl_xor(q, 2, 32);
  q += __shfl_xor(q, 4, 32);  q += __shfl_xor(q, 8, 32);
  q += __shfl_xor(q, 16, 32);
  if ((i & 31) == 0) red[i >> 5] = q;
  __syncthreads();
  float var = 0.0f;
#pragma unroll
  for (int w = 0; w < 8; ++w) var += red[w];
  var *= (1.0f / 256.0f);
  float o = d * rsqrtf(var + 1e-5f) * g[i] + beta[i];
  if (y32) y32[idx] = o;
  if (y16) y16[idx] = (_Float16)o;
}

// ---- Elementwise add / f32->f16 convert ------------------------------------
__global__ void addcvt_kernel(const float* __restrict__ a,
                              const float* __restrict__ b,
                              _Float16* __restrict__ y16,
                              float* __restrict__ y32, int n) {
  int i = blockIdx.x * blockDim.x + threadIdx.x;
  if (i < n) {
    float v = a[i] + (b ? b[i] : 0.0f);
    if (y16) y16[i] = (_Float16)v;
    if (y32) y32[i] = v;
  }
}

// ---- point/scale for the gaussian bias -------------------------------------
// point = tile8(h_w*sigmoid(ref_inter)/32) + offset ; scale = scale_pre^2
__global__ void pointscale_kernel(const float* __restrict__ offset,
                                  const float* __restrict__ scale_pre,
                                  const float* __restrict__ ref_inter,
                                  const float* __restrict__ h_w,
                                  float* __restrict__ point,
                                  float* __restrict__ scalev, int n) {
  int i = blockIdx.x * blockDim.x + threadIdx.x;
  if (i >= n) return;
  int t = i >> 4;      // token index l*B+b
  int j = i & 15;
  int b = t & (Bb_ - 1);
  int c = j & 1;
  float ri  = ref_inter[t * 2 + c];
  float sig = 1.0f / (1.0f + __expf(-ri));
  float ref = h_w[b * 2 + c] * sig * (1.0f / 32.0f);
  point[i]  = ref + offset[i];
  float sp  = scale_pre[i];
  scalev[i] = sp * sp;
}

// ---------------------------------------------------------------------------
extern "C" void kernel_launch(void* const* d_in, const int* in_sizes, int n_in,
                              void* d_out, int out_size, void* d_ws,
                              size_t ws_size, hipStream_t stream) {
  (void)in_sizes; (void)n_in; (void)out_size; (void)ws_size;
  const float* grid_in = (const float*)d_in[0];
  const float* h_w     = (const float*)d_in[1];
  const float* tgt     = (const float*)d_in[2];
  const float* memory  = (const float*)d_in[3];
  const float* pos     = (const float*)d_in[4];
  const float* qpos    = (const float*)d_in[5];
  const float* ln1_g = (const float*)d_in[6],  *ln1_b = (const float*)d_in[7];
  const float* ln2_g = (const float*)d_in[8],  *ln2_b = (const float*)d_in[9];
  const float* ln3_g = (const float*)d_in[10], *ln3_b = (const float*)d_in[11];
  const float* ln4_g = (const float*)d_in[12], *ln4_b = (const float*)d_in[13];
  const float* sa_in_w  = (const float*)d_in[14], *sa_in_b  = (const float*)d_in[15];
  const float* sa_out_w = (const float*)d_in[16], *sa_out_b = (const float*)d_in[17];
  const float* ca_in_w  = (const float*)d_in[18], *ca_in_b  = (const float*)d_in[19];
  const float* ca_out_w = (const float*)d_in[20], *ca_out_b = (const float*)d_in[21];
  const float* p1_w0 = (const float*)d_in[22], *p1_b0 = (const float*)d_in[23];
  const float* p1_w1 = (const float*)d_in[24], *p1_b1 = (const float*)d_in[25];
  const float* p1_w2 = (const float*)d_in[26], *p1_b2 = (const float*)d_in[27];
  const float* p2_w  = (const float*)d_in[28], *p2_b  = (const float*)d_in[29];
  const float* p3_w  = (const float*)d_in[30], *p3_b  = (const float*)d_in[31];
  const float* lin1_w = (const float*)d_in[32], *lin1_b = (const float*)d_in[33];
  const float* lin2_w = (const float*)d_in[34], *lin2_b = (const float*)d_in[35];
  float* out = (float*)d_out;

  // bump allocator over workspace
  char* ws = (char*)d_ws;
  size_t off = 0;
  auto alloc_h = [&](size_t n) -> _Float16* {
    _Float16* p = (_Float16*)(ws + off);
    off = (off + n * sizeof(_Float16) + 255) & ~(size_t)255;
    return p;
  };
  auto alloc_f = [&](size_t n) -> float* {
    float* p = (float*)(ws + off);
    off = (off + n * sizeof(float) + 255) & ~(size_t)255;
    return p;
  };

  // f16 weight copies
  _Float16* w_sa_in  = alloc_h(768 * 256);
  _Float16* w_sa_out = alloc_h(256 * 256);
  _Float16* w_ca_in  = alloc_h(768 * 256);
  _Float16* w_ca_out = alloc_h(256 * 256);
  _Float16* w_p1w0   = alloc_h(256 * 256);
  _Float16* w_p1w1   = alloc_h(256 * 256);
  _Float16* w_p1w2   = alloc_h(2 * 256);
  _Float16* w_p2     = alloc_h(16 * 256);
  _Float16* w_p3     = alloc_h(16 * 256);
  _Float16* w_lin1   = alloc_h(FF_ * 256);
  _Float16* w_lin2   = alloc_h(256 * FF_);
  // f16 activations
  _Float16* qk_h  = alloc_h((size_t)LB_ * Dd_);
  _Float16* tgt_h = alloc_h((size_t)LB_ * Dd_);
  _Float16* out_h = alloc_h((size_t)LB_ * Dd_);
  _Float16* h1_h  = alloc_h((size_t)LB_ * Dd_);
  _Float16* h2_h  = alloc_h((size_t)LB_ * Dd_);
  _Float16* Qsa   = alloc_h((size_t)LB_ * Dd_);
  _Float16* Ksa   = alloc_h((size_t)LB_ * Dd_);
  _Float16* Vsa   = alloc_h((size_t)LB_ * Dd_);
  _Float16* Osa   = alloc_h((size_t)LB_ * Dd_);
  _Float16* t_h   = alloc_h((size_t)LB_ * Dd_);
  _Float16* qca_h = alloc_h((size_t)LB_ * Dd_);
  _Float16* kca_h = alloc_h((size_t)SB_ * Dd_);
  _Float16* vca_h = alloc_h((size_t)SB_ * Dd_);
  _Float16* Qca   = alloc_h((size_t)LB_ * Dd_);
  _Float16* Kca   = alloc_h((size_t)SB_ * Dd_);
  _Float16* Vca   = alloc_h((size_t)SB_ * Dd_);
  _Float16* Oca   = alloc_h((size_t)LB_ * Dd_);
  _Float16* t2_h  = alloc_h((size_t)LB_ * Dd_);
  _Float16* ffn1  = alloc_h((size_t)LB_ * FF_);
  // f32 buffers
  float* out32    = alloc_f((size_t)LB_ * Dd_);
  float* sa_out32 = alloc_f((size_t)LB_ * Dd_);
  float* t32      = alloc_f((size_t)LB_ * Dd_);
  float* ca_out32 = alloc_f((size_t)LB_ * Dd_);
  float* t2_32    = alloc_f((size_t)LB_ * Dd_);
  float* ffn2_32  = alloc_f((size_t)LB_ * Dd_);
  float* offs32   = alloc_f((size_t)LB_ * 16);
  float* spre32   = alloc_f((size_t)LB_ * 16);
  float* ri32     = alloc_f((size_t)LB_ * 2);
  float* point32  = alloc_f((size_t)LB_ * 16);
  float* scale32  = alloc_f((size_t)LB_ * 16);

  auto cvt = [&](const float* src, const float* add, _Float16* dst, int n) {
    addcvt_kernel<<<(n + 255) / 256, 256, 0, stream>>>(src, add, dst, nullptr, n);
  };
  auto gemm = [&](const _Float16* A, const _Float16* Bt, const float* bias,
                  float* C32, _Float16* C16, int M, int N, int K, int relu) {
    dim3 g((N + 127) / 128, (M + 63) / 64);
    gemm_kernel<<<g, 256, 0, stream>>>(A, Bt, bias, C32, C16, M, N, K, relu);
  };

  // ---- weight conversion ----
  cvt(sa_in_w,  nullptr, w_sa_in,  768 * 256);
  cvt(sa_out_w, nullptr, w_sa_out, 256 * 256);
  cvt(ca_in_w,  nullptr, w_ca_in,  768 * 256);
  cvt(ca_out_w, nullptr, w_ca_out, 256 * 256);
  cvt(p1_w0,    nullptr, w_p1w0,   256 * 256);
  cvt(p1_w1,    nullptr, w_p1w1,   256 * 256);
  cvt(p1_w2,    nullptr, w_p1w2,   2 * 256);
  cvt(p2_w,     nullptr, w_p2,     16 * 256);
  cvt(p3_w,     nullptr, w_p3,     16 * 256);
  cvt(lin1_w,   nullptr, w_lin1,   FF_ * 256);
  cvt(lin2_w,   nullptr, w_lin2,   256 * FF_);

  // ---- inputs ----
  cvt(tgt, qpos, qk_h, LB_ * Dd_);            // q/k for self-attention
  cvt(tgt, nullptr, tgt_h, LB_ * Dd_);        // v for self-attention
  // norm4: out = LN(tgt + query_pos)
  add_ln_kernel<<<LB_, 256, 0, stream>>>(tgt, qpos, ln4_g, ln4_b, out32, out_h);

  // ---- self-attention ----
  gemm(qk_h,  w_sa_in,             sa_in_b,       nullptr, Qsa, LB_, Dd_, Dd_, 0);
  gemm(qk_h,  w_sa_in + 256 * 256, sa_in_b + 256, nullptr, Ksa, LB_, Dd_, Dd_, 0);
  gemm(tgt_h, w_sa_in + 512 * 256, sa_in_b + 512, nullptr, Vsa, LB_, Dd_, Dd_, 0);
  attn_kernel<0><<<dim3(Bb_ * Hh_, (Lq_ + 127) / 128), 256, 0, stream>>>(
      Qsa, Ksa, Vsa, Osa, nullptr, nullptr, nullptr, Lq_, Lq_);
  gemm(Osa, w_sa_out, sa_out_b, sa_out32, nullptr, LB_, Dd_, Dd_, 0);
  add_ln_kernel<<<LB_, 256, 0, stream>>>(tgt, sa_out32, ln1_g, ln1_b, t32, t_h);

  // ---- gaussian bias path (offset / scale / mlp3 ref points) ----
  gemm(out_h, w_p2,   p2_b,  offs32, nullptr, LB_, 16, Dd_, 0);
  gemm(out_h, w_p3,   p3_b,  spre32, nullptr, LB_, 16, Dd_, 0);
  gemm(out_h, w_p1w0, p1_b0, nullptr, h1_h,   LB_, Dd_, Dd_, 1);
  gemm(h1_h,  w_p1w1, p1_b1, nullptr, h2_h,   LB_, Dd_, Dd_, 1);
  gemm(h2_h,  w_p1w2, p1_b2, ri32,   nullptr, LB_, 2,   Dd_, 0);
  pointscale_kernel<<<(LB_ * 16 + 255) / 256, 256, 0, stream>>>(
      offs32, spre32, ri32, h_w, point32, scale32, LB_ * 16);

  // ---- cross-attention (bias fused into logits) ----
  cvt(t32,    qpos,    qca_h, LB_ * Dd_);
  cvt(memory, pos,     kca_h, SB_ * Dd_);
  cvt(memory, nullptr, vca_h, SB_ * Dd_);
  gemm(qca_h, w_ca_in,             ca_in_b,       nullptr, Qca, LB_, Dd_, Dd_, 0);
  gemm(kca_h, w_ca_in + 256 * 256, ca_in_b + 256, nullptr, Kca, SB_, Dd_, Dd_, 0);
  gemm(vca_h, w_ca_in + 512 * 256, ca_in_b + 512, nullptr, Vca, SB_, Dd_, Dd_, 0);
  attn_kernel<1><<<dim3(Bb_ * Hh_, (Lq_ + 127) / 128), 256, 0, stream>>>(
      Qca, Kca, Vca, Oca, point32, scale32, grid_in, Lq_, Ss_);
  gemm(Oca, w_ca_out, ca_out_b, ca_out32, nullptr, LB_, Dd_, Dd_, 0);
  add_ln_kernel<<<LB_, 256, 0, stream>>>(t32, ca_out32, ln2_g, ln2_b, t2_32, t2_h);

  // ---- FFN ----
  gemm(t2_h, w_lin1, lin1_b, nullptr, ffn1,    LB_, FF_, Dd_, 1);
  gemm(ffn1, w_lin2, lin2_b, ffn2_32, nullptr, LB_, Dd_, FF_, 0);
  add_ln_kernel<<<LB_, 256, 0, stream>>>(t2_32, ffn2_32, ln3_g, ln3_b, out, nullptr);
}
